// Attention_6940667150551
// MI455X (gfx1250) — matrix operand order repaired and checked by reading, not compiled, not run on hardware
//
#include <hip/hip_runtime.h>
#include <math.h>

#define DIM 512
#define NH 8
#define HD 64
#define AG_ 144
#define WIN 18
#define PS 12
#define BB 128
#define N0 320
#define NTOK 324   // reference padded token count (18*18)
#define NP 352     // WMMA-padded tokens (multiple of 32)
#define NP2 336    // M-tile padded tokens for proj (multiple of 16)
#define AGP 160    // WMMA-padded agent count (multiple of 32)

typedef __bf16 bf16;
typedef __attribute__((ext_vector_type(16))) __bf16 v16bf;
typedef __attribute__((ext_vector_type(8)))  float  v8f;

// ---------------- WMMA fragment loaders (contiguous k order, wave32) ------------

// A fragment 16x32 (MxK), source row-major [m][k] with row stride `stride`.
// lanes 0-15: M=lane, K=0..15 ; lanes 16-31: M=lane-16, K=16..31
static __device__ inline v16bf load_a_rowmajor(const bf16* tile, int stride, int lane) {
  int m  = lane & 15;
  int kb = (lane & 16) ? 16 : 0;
  const bf16* p = tile + (size_t)m * stride + kb;
  v16bf a;
#pragma unroll
  for (int e = 0; e < 16; ++e) a[e] = p[e];
  return a;
}

// B fragment 32x16 (KxN), source laid out [n][k] (k contiguous, row stride `stride`).
// lanes 0-15: N=lane, K=0..15 ; lanes 16-31: N=lane-16, K=16..31
static __device__ inline v16bf load_b_kcontig(const bf16* tile, int stride, int lane) {
  int n  = lane & 15;
  int kb = (lane & 16) ? 16 : 0;
  const bf16* p = tile + (size_t)n * stride + kb;
  v16bf b;
#pragma unroll
  for (int e = 0; e < 16; ++e) b[e] = p[e];
  return b;
}

static __device__ inline v8f wmma_bf16(v16bf a, v16bf b, v8f c) {
  return __builtin_amdgcn_wmma_f32_16x16x32_bf16(false, a, false, b, (short)0, c, false, false);
}

// ---------------- prep kernels ---------------------------------------------------

__global__ void k_transpose_bf16(const float* __restrict__ w, bf16* __restrict__ wt,
                                 int K, int N) {
  int idx = blockIdx.x * blockDim.x + threadIdx.x;
  if (idx >= K * N) return;
  int n = idx % N, k = idx / N;
  wt[(size_t)n * K + k] = (bf16)w[(size_t)k * N + n];
}

__global__ void k_build_x(const float* __restrict__ x, bf16* __restrict__ xb) {
  int idx = blockIdx.x * blockDim.x + threadIdx.x;
  if (idx >= BB * NP * DIM) return;
  int c = idx % DIM;
  int t = (idx / DIM) % NP;
  int b = idx / (DIM * NP);
  float v = 0.f;
  if (t >= 2 && t < 2 + N0) v = x[((size_t)b * N0 + (t - 2)) * DIM + c];
  xb[idx] = (bf16)v;
}

__global__ void k_zero_bf16(bf16* __restrict__ p, int total) {
  int idx = blockIdx.x * blockDim.x + threadIdx.x;
  if (idx >= total) return;
  p[idx] = (bf16)0.f;
}

static __device__ inline float bilerp7(const float* m, int oi, int oj) {
  const float sc = 7.0f / 18.0f;            // half-pixel bilinear 7x7 -> 18x18
  float y = (oi + 0.5f) * sc - 0.5f;
  float x = (oj + 0.5f) * sc - 0.5f;
  float yf = floorf(y), xf = floorf(x);
  float fy = y - yf, fx = x - xf;
  int y0 = (int)yf, x0 = (int)xf;
  int y0c = min(max(y0, 0), 6), y1c = min(max(y0 + 1, 0), 6);
  int x0c = min(max(x0, 0), 6), x1c = min(max(x0 + 1, 0), 6);
  float v00 = m[y0c * 7 + x0c], v01 = m[y0c * 7 + x1c];
  float v10 = m[y1c * 7 + x0c], v11 = m[y1c * 7 + x1c];
  return (1.f - fy) * ((1.f - fx) * v00 + fx * v01) + fy * ((1.f - fx) * v10 + fx * v11);
}

// pb[h][a][n] : resize(an_bias) + ah + aw ; padded columns -> -1e30
__global__ void k_bias_pb(const float* __restrict__ an, const float* __restrict__ ah,
                          const float* __restrict__ aw, float* __restrict__ pb) {
  int idx = blockIdx.x * blockDim.x + threadIdx.x;
  if (idx >= NH * AG_ * NP) return;
  int n = idx % NP;
  int a = (idx / NP) % AG_;
  int h = idx / (NP * AG_);
  float v;
  if (n >= NTOK) v = -1e30f;
  else {
    int i = n / WIN, j = n % WIN;
    v = bilerp7(an + ((size_t)h * AG_ + a) * 49, i, j)
      + ah[((size_t)h * AG_ + a) * WIN + i]
      + aw[((size_t)h * AG_ + a) * WIN + j];
  }
  pb[idx] = v;
}

// ab[h][n][a] : resize(na_bias)^T + ha + wa ; padded rows -> 0
__global__ void k_bias_ab(const float* __restrict__ na, const float* __restrict__ ha,
                          const float* __restrict__ wa, float* __restrict__ ab) {
  int idx = blockIdx.x * blockDim.x + threadIdx.x;
  if (idx >= NH * NP * AG_) return;
  int a = idx % AG_;
  int n = (idx / AG_) % NP;
  int h = idx / (AG_ * NP);
  float v = 0.f;
  if (n < NTOK) {
    int i = n / WIN, j = n % WIN;
    v = bilerp7(na + ((size_t)h * AG_ + a) * 49, i, j)
      + ha[((size_t)h * WIN + i) * AG_ + a]
      + wa[((size_t)h * WIN + j) * AG_ + a];
  }
  ab[idx] = v;
}

// ---------------- qkv GEMM: (B*NP x 512) @ (512 x 1536) ---------------------------
// One wave computes a 16x64 strip: 1 A fragment reused across 4 B fragments.
// All fragment loads precede the WMMA chain so loads overlap matrix math.

__global__ void __launch_bounds__(256) k_qkv(const bf16* __restrict__ xb,
                                             const bf16* __restrict__ wt,
                                             bf16* __restrict__ q, bf16* __restrict__ kk,
                                             bf16* __restrict__ vv, bf16* __restrict__ vt) {
  const int MT = NP / 16;          // 22
  const int NG = (3 * DIM) / 64;   // 24 groups of 4 n-tiles
  int wave = threadIdx.x >> 5, lane = threadIdx.x & 31;
  int gw = blockIdx.x * 8 + wave;
  if (gw >= BB * MT * NG) return;
  int ng = gw % NG;
  int mt = (gw / NG) % MT;
  int b  = gw / (NG * MT);
  const bf16* abase = xb + ((size_t)b * NP + mt * 16) * DIM;
  const bf16* bbase = wt + (size_t)(ng * 64) * DIM;
  v8f acc[4] = {{}, {}, {}, {}};
  for (int ks = 0; ks < DIM; ks += 32) {
    v16bf a  = load_a_rowmajor(abase + ks, DIM, lane);
    v16bf b0 = load_b_kcontig(bbase + ks, DIM, lane);
    v16bf b1 = load_b_kcontig(bbase + (size_t)16 * DIM + ks, DIM, lane);
    v16bf b2 = load_b_kcontig(bbase + (size_t)32 * DIM + ks, DIM, lane);
    v16bf b3 = load_b_kcontig(bbase + (size_t)48 * DIM + ks, DIM, lane);
    acc[0] = wmma_bf16(a, b0, acc[0]);
    acc[1] = wmma_bf16(a, b1, acc[1]);
    acc[2] = wmma_bf16(a, b2, acc[2]);
    acc[3] = wmma_bf16(a, b3, acc[3]);
  }
  int which = (ng * 64) / DIM;          // uniform per group: 0=q 1=k 2=v
  int cbase = (ng * 64) % DIM;          // multiple of 64 -> single head per group
  bf16* dst = (which == 0) ? q : ((which == 1) ? kk : vv);
  int h = cbase >> 6;
  int n = lane & 15, mof = (lane & 16) ? 8 : 0;
  size_t bh = (size_t)b * NH + h;
#pragma unroll
  for (int t = 0; t < 4; ++t) {
    int d = t * 16 + n;
#pragma unroll
    for (int r = 0; r < 8; ++r) {
      int tok = mt * 16 + mof + r;
      bf16 val = (bf16)acc[t][r];
      dst[(bh * NP + tok) * HD + d] = val;
      if (which == 2) vt[(bh * HD + d) * NP + tok] = val;   // transposed V for PV GEMM
    }
  }
}

// ---------------- agent pooling (adaptive avg pool of q, 18x18 -> 12x12) ---------

__global__ void k_agent_pool(const bf16* __restrict__ q, bf16* __restrict__ ag) {
  int idx = blockIdx.x * blockDim.x + threadIdx.x;
  if (idx >= BB * NH * AG_ * HD) return;
  int d = idx & 63;
  int a = (idx >> 6) % AG_;
  int h = (idx / (64 * AG_)) % NH;
  int b = idx / (64 * AG_ * NH);
  int oi = a / PS, oj = a % PS;
  int si = (oi * WIN) / PS, ei = ((oi + 1) * WIN + PS - 1) / PS;
  int sj = (oj * WIN) / PS, ej = ((oj + 1) * WIN + PS - 1) / PS;
  const bf16* qp = q + (((size_t)b * NH + h) * NP) * HD + d;
  float s = 0.f;
  for (int i = si; i < ei; ++i)
    for (int j = sj; j < ej; ++j)
      s += (float)qp[(size_t)(i * WIN + j) * HD];
  s /= (float)((ei - si) * (ej - sj));
  ag[idx] = (bf16)s;
}

// ---------------- stage 1: agent_attn = softmax(scale*ag@k^T + pb); avT = (attn@v)^T

__global__ void __launch_bounds__(256) k_stage1(const bf16* __restrict__ ag,
                                                const bf16* __restrict__ kk,
                                                const bf16* __restrict__ vt,
                                                const float* __restrict__ pb,
                                                bf16* __restrict__ avt) {
  __shared__ float S[16 * NP];
  __shared__ bf16  P[16 * NP];
  int wg = blockIdx.x;
  int at = wg % (AG_ / 16);
  int h  = (wg / (AG_ / 16)) % NH;
  int b  = wg / ((AG_ / 16) * NH);
  int tid = threadIdx.x, wave = tid >> 5, lane = tid & 31;
  const float scale = 0.125f;  // 64^-0.5
  size_t bh = (size_t)b * NH + h;
  const bf16* agp = ag + (bh * AG_ + at * 16) * HD;
  const bf16* kp  = kk + bh * NP * HD;
  const bf16* vtp = vt + bh * HD * NP;
  const float* pbp = pb + ((size_t)h * AG_ + at * 16) * NP;

  // A fragments (16x64 agent tile) are invariant across column tiles: load once.
  v16bf a0 = load_a_rowmajor(agp, HD, lane);
  v16bf a1 = load_a_rowmajor(agp + 32, HD, lane);
  for (int ct = wave; ct < NP / 16; ct += 8) {
    v16bf b0 = load_b_kcontig(kp + (size_t)(ct * 16) * HD, HD, lane);
    v16bf b1 = load_b_kcontig(kp + (size_t)(ct * 16) * HD + 32, HD, lane);
    v8f acc = {};
    acc = wmma_bf16(a0, b0, acc);
    acc = wmma_bf16(a1, b1, acc);
    int n = lane & 15, mof = (lane & 16) ? 8 : 0;
#pragma unroll
    for (int r = 0; r < 8; ++r) {
      int m = mof + r, col = ct * 16 + n;
      S[m * NP + col] = acc[r] * scale + pbp[(size_t)m * NP + col];
    }
  }
  __syncthreads();
  {   // row softmax: 16 threads per row, shfl_xor reductions
    int row = tid >> 4, sub = tid & 15;
    float mx = -1e30f;
    for (int c2 = sub; c2 < NP; c2 += 16) mx = fmaxf(mx, S[row * NP + c2]);
    for (int m2 = 8; m2 >= 1; m2 >>= 1) mx = fmaxf(mx, __shfl_xor(mx, m2, 16));
    float sum = 0.f;
    for (int c2 = sub; c2 < NP; c2 += 16) {
      float e = __expf(S[row * NP + c2] - mx);
      S[row * NP + c2] = e; sum += e;
    }
    for (int m2 = 8; m2 >= 1; m2 >>= 1) sum += __shfl_xor(sum, m2, 16);
    float inv = 1.f / sum;
    for (int c2 = sub; c2 < NP; c2 += 16) P[row * NP + c2] = (bf16)(S[row * NP + c2] * inv);
  }
  __syncthreads();
  if (wave < 4) {
    int ct = wave;                       // 4 col tiles over HD
    v8f acc = {};
    for (int ks = 0; ks < NP; ks += 32) {
      v16bf a  = load_a_rowmajor(P + ks, NP, lane);
      v16bf bm = load_b_kcontig(vtp + (size_t)(ct * 16) * NP + ks, NP, lane);
      acc = wmma_bf16(a, bm, acc);
    }
    int n = lane & 15, mof = (lane & 16) ? 8 : 0;
    int d = ct * 16 + n;
#pragma unroll
    for (int r = 0; r < 8; ++r) {
      int arow = at * 16 + mof + r;
      avt[(bh * HD + d) * AGP + arow] = (bf16)acc[r];   // transposed agent_v
    }
  }
}

// ---------------- stage 2: q_attn = softmax(scale*q@ag^T + ab); out = attn@av -----

__global__ void __launch_bounds__(256) k_stage2(const bf16* __restrict__ q,
                                                const bf16* __restrict__ ag,
                                                const bf16* __restrict__ avt,
                                                const float* __restrict__ ab,
                                                float* __restrict__ o1) {
  __shared__ float S[16 * AGP];
  __shared__ bf16  P[16 * AGP];
  int wg = blockIdx.x;
  int nt = wg % (NP / 16);
  int h  = (wg / (NP / 16)) % NH;
  int b  = wg / ((NP / 16) * NH);
  int tid = threadIdx.x, wave = tid >> 5, lane = tid & 31;
  const float scale = 0.125f;
  size_t bh = (size_t)b * NH + h;
  const bf16* qp   = q   + (bh * NP + nt * 16) * HD;
  const bf16* agp  = ag  + bh * AG_ * HD;
  const bf16* avtp = avt + bh * HD * AGP;
  const float* abp = ab + ((size_t)h * NP + nt * 16) * AG_;

  v16bf a0 = load_a_rowmajor(qp, HD, lane);
  v16bf a1 = load_a_rowmajor(qp + 32, HD, lane);
  for (int ct = wave; ct < AG_ / 16; ct += 8) {   // 9 tiles over agents
    v16bf b0 = load_b_kcontig(agp + (size_t)(ct * 16) * HD, HD, lane);
    v16bf b1 = load_b_kcontig(agp + (size_t)(ct * 16) * HD + 32, HD, lane);
    v8f acc = {};
    acc = wmma_bf16(a0, b0, acc);
    acc = wmma_bf16(a1, b1, acc);
    int n = lane & 15, mof = (lane & 16) ? 8 : 0;
#pragma unroll
    for (int r = 0; r < 8; ++r) {
      int m = mof + r, col = ct * 16 + n;
      S[m * AGP + col] = acc[r] * scale + abp[(size_t)m * AG_ + col];
    }
  }
  __syncthreads();
  {
    int row = tid >> 4, sub = tid & 15;
    float mx = -1e30f;
    for (int c2 = sub; c2 < AG_; c2 += 16) mx = fmaxf(mx, S[row * AGP + c2]);
    for (int m2 = 8; m2 >= 1; m2 >>= 1) mx = fmaxf(mx, __shfl_xor(mx, m2, 16));
    float sum = 0.f;
    for (int c2 = sub; c2 < AG_; c2 += 16) {
      float e = __expf(S[row * AGP + c2] - mx);
      S[row * AGP + c2] = e; sum += e;
    }
    for (int m2 = 8; m2 >= 1; m2 >>= 1) sum += __shfl_xor(sum, m2, 16);
    float inv = 1.f / sum;
    for (int c2 = sub; c2 < AG_; c2 += 16) P[row * AGP + c2] = (bf16)(S[row * AGP + c2] * inv);
    for (int c2 = AG_ + sub; c2 < AGP; c2 += 16) P[row * AGP + c2] = (bf16)0.f;
  }
  __syncthreads();
  if (wave < 4) {
    int ct = wave;
    v8f acc = {};
    for (int ks = 0; ks < AGP; ks += 32) {       // K padded 144 -> 160 (zeros both sides)
      v16bf a  = load_a_rowmajor(P + ks, AGP, lane);
      v16bf bm = load_b_kcontig(avtp + (size_t)(ct * 16) * AGP + ks, AGP, lane);
      acc = wmma_bf16(a, bm, acc);
    }
    int n = lane & 15, mof = (lane & 16) ? 8 : 0;
#pragma unroll
    for (int r = 0; r < 8; ++r) {
      int tok = nt * 16 + mof + r;
      if (tok < NP2) {
        float val = (tok < NTOK) ? acc[r] : 0.f;
        o1[((size_t)b * NP2 + tok) * DIM + h * HD + ct * 16 + n] = val;
      }
    }
  }
}

// ---------------- depthwise 3x3 conv on v image, added into o1 -------------------

__global__ void k_dwc(const bf16* __restrict__ vv, const float* __restrict__ dw,
                      const float* __restrict__ db, float* __restrict__ o1) {
  int idx = blockIdx.x * blockDim.x + threadIdx.x;
  if (idx >= BB * NTOK * DIM) return;
  int c   = idx % DIM;
  int tok = (idx / DIM) % NTOK;
  int b   = idx / (DIM * NTOK);
  int i = tok / WIN, j = tok % WIN;
  int h = c >> 6, d = c & 63;
  const bf16* vp = vv + (((size_t)b * NH + h) * NP) * HD + d;
  float s = db[c];
#pragma unroll
  for (int di = 0; di < 3; ++di) {
    int ii = i + di - 1;
    if (ii < 0 || ii >= WIN) continue;
#pragma unroll
    for (int dj = 0; dj < 3; ++dj) {
      int jj = j + dj - 1;
      if (jj < 0 || jj >= WIN) continue;
      s += (float)vp[(size_t)(ii * WIN + jj) * HD] * dw[c * 9 + di * 3 + dj];
    }
  }
  o1[((size_t)b * NP2 + tok) * DIM + c] += s;
}

__global__ void k_f2b(const float* __restrict__ src, bf16* __restrict__ dst, int total) {
  int idx = blockIdx.x * blockDim.x + threadIdx.x;
  if (idx >= total) return;
  dst[idx] = (bf16)src[idx];
}

// ---------------- proj GEMM + bias + token slice ---------------------------------

__global__ void __launch_bounds__(256) k_proj(const bf16* __restrict__ o1b,
                                              const bf16* __restrict__ wt,
                                              const float* __restrict__ pbias,
                                              float* __restrict__ out) {
  const int MT = NP2 / 16;   // 21
  const int NG = DIM / 64;   // 8 groups of 4 n-tiles
  int wave = threadIdx.x >> 5, lane = threadIdx.x & 31;
  int gw = blockIdx.x * 8 + wave;
  if (gw >= BB * MT * NG) return;
  int ng = gw % NG;
  int mt = (gw / NG) % MT;
  int b  = gw / (NG * MT);
  const bf16* abase = o1b + ((size_t)b * NP2 + mt * 16) * DIM;
  const bf16* bbase = wt + (size_t)(ng * 64) * DIM;
  v8f acc[4] = {{}, {}, {}, {}};
  for (int ks = 0; ks < DIM; ks += 32) {
    v16bf a  = load_a_rowmajor(abase + ks, DIM, lane);
    v16bf b0 = load_b_kcontig(bbase + ks, DIM, lane);
    v16bf b1 = load_b_kcontig(bbase + (size_t)16 * DIM + ks, DIM, lane);
    v16bf b2 = load_b_kcontig(bbase + (size_t)32 * DIM + ks, DIM, lane);
    v16bf b3 = load_b_kcontig(bbase + (size_t)48 * DIM + ks, DIM, lane);
    acc[0] = wmma_bf16(a, b0, acc[0]);
    acc[1] = wmma_bf16(a, b1, acc[1]);
    acc[2] = wmma_bf16(a, b2, acc[2]);
    acc[3] = wmma_bf16(a, b3, acc[3]);
  }
  int n = lane & 15, mof = (lane & 16) ? 8 : 0;
#pragma unroll
  for (int t = 0; t < 4; ++t) {
    int col = ng * 64 + t * 16 + n;
    float bias = pbias[col];
#pragma unroll
    for (int r = 0; r < 8; ++r) {
      int tok = mt * 16 + mof + r;
      if (tok >= 2 && tok < 2 + N0)
        out[((size_t)b * N0 + (tok - 2)) * DIM + col] = acc[t][r] + bias;
    }
  }
}

// ---------------- launch ---------------------------------------------------------

static inline int cdiv(long long a, int b) { return (int)((a + b - 1) / b); }

extern "C" void kernel_launch(void* const* d_in, const int* in_sizes, int n_in,
                              void* d_out, int out_size, void* d_ws, size_t ws_size,
                              hipStream_t stream) {
  (void)in_sizes; (void)n_in; (void)out_size; (void)ws_size;
  const float* x      = (const float*)d_in[0];
  const float* qkv_w  = (const float*)d_in[1];
  const float* proj_w = (const float*)d_in[2];
  const float* proj_b = (const float*)d_in[3];
  const float* dwc_w  = (const float*)d_in[4];
  const float* dwc_b  = (const float*)d_in[5];
  const float* an_b   = (const float*)d_in[6];
  const float* na_b   = (const float*)d_in[7];
  const float* ah_b   = (const float*)d_in[8];
  const float* aw_b   = (const float*)d_in[9];
  const float* ha_b   = (const float*)d_in[10];
  const float* wa_b   = (const float*)d_in[11];
  float* out = (float*)d_out;

  char* ws = (char*)d_ws;
  size_t off = 0;
  auto alloc = [&](size_t bytes) -> void* {
    void* p = ws + off;
    off += (bytes + 255) & ~(size_t)255;
    return p;
  };
  bf16* xb     = (bf16*)alloc((size_t)BB * NP * DIM * 2);
  bf16* wqkvT  = (bf16*)alloc((size_t)3 * DIM * DIM * 2);
  bf16* wprojT = (bf16*)alloc((size_t)DIM * DIM * 2);
  bf16* qb     = (bf16*)alloc((size_t)BB * NH * NP * HD * 2);
  bf16* kbuf   = (bf16*)alloc((size_t)BB * NH * NP * HD * 2);
  bf16* vbuf   = (bf16*)alloc((size_t)BB * NH * NP * HD * 2);
  bf16* vtbuf  = (bf16*)alloc((size_t)BB * NH * HD * NP * 2);
  bf16* agb    = (bf16*)alloc((size_t)BB * NH * AG_ * HD * 2);
  float* pbuf  = (float*)alloc((size_t)NH * AG_ * NP * 4);
  float* abuf  = (float*)alloc((size_t)NH * NP * AG_ * 4);
  bf16* avtb   = (bf16*)alloc((size_t)BB * NH * HD * AGP * 2);
  float* o1    = (float*)alloc((size_t)BB * NP2 * DIM * 4);
  bf16* o1b    = (bf16*)alloc((size_t)BB * NP2 * DIM * 2);

  // prep
  k_transpose_bf16<<<cdiv((long long)DIM * 3 * DIM, 256), 256, 0, stream>>>(qkv_w, wqkvT, DIM, 3 * DIM);
  k_transpose_bf16<<<cdiv((long long)DIM * DIM, 256), 256, 0, stream>>>(proj_w, wprojT, DIM, DIM);
  k_build_x<<<cdiv((long long)BB * NP * DIM, 256), 256, 0, stream>>>(x, xb);
  k_bias_pb<<<cdiv((long long)NH * AG_ * NP, 256), 256, 0, stream>>>(an_b, ah_b, aw_b, pbuf);
  k_bias_ab<<<cdiv((long long)NH * NP * AG_, 256), 256, 0, stream>>>(na_b, ha_b, wa_b, abuf);
  // zero avT so its padded agent rows (144..159) contribute nothing in stage 2
  k_zero_bf16<<<cdiv((long long)BB * NH * HD * AGP, 256), 256, 0, stream>>>(avtb, BB * NH * HD * AGP);

  // qkv GEMM (WMMA, 16x64 per wave; also emits transposed V)
  k_qkv<<<BB * (NP / 16) * ((3 * DIM) / 64) / 8, 256, 0, stream>>>(xb, wqkvT, qb, kbuf, vbuf, vtbuf);

  // agent pooling
  k_agent_pool<<<cdiv((long long)BB * NH * AG_ * HD, 256), 256, 0, stream>>>(qb, agb);

  // attention stages (fused WMMA + softmax)
  k_stage1<<<BB * NH * (AG_ / 16), 256, 0, stream>>>(agb, kbuf, vtbuf, pbuf, avtb);
  k_stage2<<<BB * NH * (NP / 16), 256, 0, stream>>>(qb, agb, avtb, abuf, o1);

  // depthwise conv skip + projection
  k_dwc<<<cdiv((long long)BB * NTOK * DIM, 256), 256, 0, stream>>>(vbuf, dwc_w, dwc_b, o1);
  k_f2b<<<cdiv((long long)BB * NP2 * DIM, 256), 256, 0, stream>>>(o1, o1b, BB * NP2 * DIM);
  k_proj<<<BB * (NP2 / 16) * (DIM / 64) / 8, 256, 0, stream>>>(o1b, wprojT, proj_b, out);
}
